// MambaEncoder_10711648436254
// MI455X (gfx1250) — matrix operand order repaired and checked by reading, not compiled, not run on hardware
//
#include <hip/hip_runtime.h>
#include <hip/hip_bf16.h>
#include <math.h>

// ---------------------------------------------------------------------------
// MambaEncoder forward for MI455X (gfx1250, wave32, WMMA).
// GEMM-shaped ops (projections + every 1x1x1 conv) use v_wmma_f32_16x16x32_f16
// with: A tile staged into LDS via global_load_async_to_lds_b32 (ASYNCcnt) and
// read back as ds_load_b128; B read as global_load_b128; branchless edge
// handling (clamp+select) so EXEC is all-ones at every WMMA.
// ---------------------------------------------------------------------------

typedef __attribute__((ext_vector_type(16))) _Float16 v16h;
typedef __attribute__((ext_vector_type(8)))  float    v8f;

#define ACT_NONE 0
#define ACT_RELU 1
#define ACT_SIGMOID 2
#define ACT_SILU 3
#define ACT_GELU 4
#define ACT_SOFTPLUS 5

__device__ __forceinline__ float act_apply(float x, int act) {
  switch (act) {
    case ACT_RELU:     return x > 0.f ? x : 0.f;
    case ACT_SIGMOID:  return 1.f / (1.f + __expf(-x));
    case ACT_SILU:     return x / (1.f + __expf(-x));
    case ACT_GELU:     return 0.5f * x * (1.f + erff(x * 0.7071067811865475f));
    case ACT_SOFTPLUS: return x > 20.f ? x : log1pf(__expf(x));
    default:           return x;
  }
}

// ---------------------------------------------------------------------------
// Generic WMMA GEMM:  C[M,N] = act( A * W^T + bias ) + res
//   A: a_cm==0 -> A[m*lda + k] ; a_cm==1 -> A[k*lda + m]
//   W: (N x K) row-major, W[n*ldw + k]
//   C: c_cm==0 -> C[m*ldc + n] ; c_cm==1 -> C[n*ldc + m]
// One wave -> one 16x16 tile; 8 waves/block share one A tile staged in LDS.
// ---------------------------------------------------------------------------
__global__ void __launch_bounds__(256)
wmma_gemm_kernel(const float* __restrict__ A, int lda, int a_cm,
                 const float* __restrict__ W, int ldw,
                 const float* __restrict__ bias,
                 float* __restrict__ C, int ldc, int c_cm,
                 const float* __restrict__ res,
                 int M, int N, int K, int act) {
  __shared__ float ldsA[16 * 36];  // 16 rows x 32 K, padded to 36 (16B-aligned runs)
  const int tid = threadIdx.x;
  const int wave = tid >> 5;
  const int lane = tid & 31;
  const int m0 = blockIdx.x * 16;
  const int n0 = (blockIdx.y * 8 + wave) * 16;
  const int lg = lane >> 4;
  const int mr = lane & 15;
  const int n = n0 + mr;
  const bool wave_on = (n0 < N);                                // wave-uniform
  const bool n_interior = (n0 + 16 <= N) && ((ldw & 3) == 0);   // wave-uniform

  v8f acc = {0.f, 0.f, 0.f, 0.f, 0.f, 0.f, 0.f, 0.f};

  for (int k0 = 0; k0 < K; k0 += 32) {
    const bool kfull = (k0 + 32 <= K);            // block-uniform
    const bool a_async = kfull && (m0 + 16 <= M); // block-uniform

    // ---- stage A tile (16x32 f32) into LDS: 2 elements per thread --------
    if (a_async) {
      // CDNA5 async DMA into LDS; LDS dest offset = static-LDS byte offset.
#pragma unroll
      for (int j = 0; j < 2; ++j) {
        int e = tid * 2 + j;
        int r = e >> 5, kk = e & 31;
        size_t gix = a_cm ? ((size_t)(k0 + kk) * lda + (m0 + r))
                          : ((size_t)(m0 + r) * lda + (k0 + kk));
        unsigned ldst = (unsigned)((r * 36 + kk) * 4);
        asm volatile("global_load_async_to_lds_b32 %0, %1, off"
                     :: "v"(ldst), "v"(A + gix)
                     : "memory");
      }
    } else {
      // edge tile: clamped, branchless staging with zero fill
#pragma unroll
      for (int j = 0; j < 2; ++j) {
        int e = tid * 2 + j;
        int r = e >> 5, kk = e & 31;
        int m = m0 + r, ka = k0 + kk;
        bool ok = (m < M) && (ka < K);
        int mc = m < M ? m : M - 1;
        int kc = ka < K ? ka : K - 1;
        float v = A[a_cm ? ((size_t)kc * lda + mc) : ((size_t)mc * lda + kc)];
        ldsA[r * 36 + kk] = ok ? v : 0.f;
      }
    }
    asm volatile("s_wait_asynccnt 0x0" ::: "memory");
    __syncthreads();

    if (wave_on) {
      // ---- A fragment: one 64B run of 16 contiguous K per lane from LDS (ds_load_b128) ----
      const float* rp = &ldsA[mr * 36 + lg * 16];
      float4 f0 = *(const float4*)(rp);
      float4 f1 = *(const float4*)(rp + 4);
      float4 f2 = *(const float4*)(rp + 8);
      float4 f3 = *(const float4*)(rp + 12);
      v16h a;
      a[0]  = (_Float16)f0.x; a[1]  = (_Float16)f0.y;
      a[2]  = (_Float16)f0.z; a[3]  = (_Float16)f0.w;
      a[4]  = (_Float16)f1.x; a[5]  = (_Float16)f1.y;
      a[6]  = (_Float16)f1.z; a[7]  = (_Float16)f1.w;
      a[8]  = (_Float16)f2.x; a[9]  = (_Float16)f2.y;
      a[10] = (_Float16)f2.z; a[11] = (_Float16)f2.w;
      a[12] = (_Float16)f3.x; a[13] = (_Float16)f3.y;
      a[14] = (_Float16)f3.z; a[15] = (_Float16)f3.w;

      // ---- B fragment: 16 contiguous K floats of row n -------------------
      v16h b;
      if (n_interior && kfull) {
        const float* wp2 = W + (size_t)n * ldw + k0 + lg * 16;
        __builtin_prefetch(wp2 + 32, 0, 0);  // next K-slab (global_prefetch_b8)
        float4 b0 = ((const float4*)wp2)[0];
        float4 b1 = ((const float4*)wp2)[1];
        float4 b2 = ((const float4*)wp2)[2];
        float4 b3 = ((const float4*)wp2)[3];
        b[0]  = (_Float16)b0.x; b[1]  = (_Float16)b0.y;
        b[2]  = (_Float16)b0.z; b[3]  = (_Float16)b0.w;
        b[4]  = (_Float16)b1.x; b[5]  = (_Float16)b1.y;
        b[6]  = (_Float16)b1.z; b[7]  = (_Float16)b1.w;
        b[8]  = (_Float16)b2.x; b[9]  = (_Float16)b2.y;
        b[10] = (_Float16)b2.z; b[11] = (_Float16)b2.w;
        b[12] = (_Float16)b3.x; b[13] = (_Float16)b3.y;
        b[14] = (_Float16)b3.z; b[15] = (_Float16)b3.w;
      } else {
#pragma unroll
        for (int i = 0; i < 16; ++i) {
          int kb = k0 + i + lg * 16;
          int nc2 = n < N ? n : N - 1;
          int kc2 = kb < K ? kb : K - 1;
          float bv = W[(size_t)nc2 * ldw + kc2];   // unconditional load
          b[i] = (n < N && kb < K) ? (_Float16)bv : (_Float16)0.f;
        }
      }
      acc = __builtin_amdgcn_wmma_f32_16x16x32_f16(false, a, false, b,
                                                   (short)0, acc, false, false);
    }
    __syncthreads();
  }

  if (wave_on) {
#pragma unroll
    for (int r = 0; r < 8; ++r) {
      int m = m0 + r + lg * 8;
      if (m < M && n < N) {
        float v = acc[r];
        if (bias) v += bias[n];
        v = act_apply(v, act);
        size_t o = c_cm ? ((size_t)n * ldc + m) : ((size_t)m * ldc + n);
        if (res) v += res[o];
        C[o] = v;
      }
    }
  }
}

// ---------------------------------------------------------------------------
// Direct 3D conv (cubic kernel K), NCDHW / OIDHW. One thread per output elem.
// ---------------------------------------------------------------------------
__global__ void conv3d_kernel(const float* __restrict__ x,
                              const float* __restrict__ w,
                              const float* __restrict__ bias,
                              float* __restrict__ y,
                              int N, int Ci, int ID, int IH, int IW,
                              int Co, int K, int stride, int pad,
                              int OD, int OH, int OW) {
  size_t idx = (size_t)blockIdx.x * blockDim.x + threadIdx.x;
  size_t total = (size_t)N * Co * OD * OH * OW;
  if (idx >= total) return;
  int ow = idx % OW; size_t t = idx / OW;
  int oh = t % OH; t /= OH;
  int od = t % OD; t /= OD;
  int co = t % Co; int n = (int)(t / Co);
  float acc = bias ? bias[co] : 0.f;
  int d0 = od * stride - pad, h0 = oh * stride - pad, w0 = ow * stride - pad;
  for (int ci = 0; ci < Ci; ++ci) {
    const float* xp = x + ((size_t)n * Ci + ci) * ID * IH * IW;
    const float* wp = w + ((size_t)co * Ci + ci) * K * K * K;
    for (int kd = 0; kd < K; ++kd) {
      int dz = d0 + kd; if (dz < 0 || dz >= ID) continue;
      for (int kh = 0; kh < K; ++kh) {
        int hy = h0 + kh; if (hy < 0 || hy >= IH) continue;
        for (int kw = 0; kw < K; ++kw) {
          int wx = w0 + kw; if (wx < 0 || wx >= IW) continue;
          acc += xp[((size_t)dz * IH + hy) * IW + wx] * wp[(kd * K + kh) * K + kw];
        }
      }
    }
  }
  y[idx] = acc;
}

// ---------------------------------------------------------------------------
// Instance norm: stats per (n,c) then normalize (+ optional activation).
// ---------------------------------------------------------------------------
__global__ void inorm_stats_kernel(const float* __restrict__ x,
                                   float* __restrict__ stats, int Lsz) {
  int nc = blockIdx.x;
  const float* p = x + (size_t)nc * Lsz;
  float s = 0.f, s2 = 0.f;
  for (int i = threadIdx.x; i < Lsz; i += blockDim.x) {
    float v = p[i]; s += v; s2 += v * v;
  }
  __shared__ float sh0[256], sh1[256];
  sh0[threadIdx.x] = s; sh1[threadIdx.x] = s2;
  __syncthreads();
  for (int off = 128; off > 0; off >>= 1) {
    if ((int)threadIdx.x < off) {
      sh0[threadIdx.x] += sh0[threadIdx.x + off];
      sh1[threadIdx.x] += sh1[threadIdx.x + off];
    }
    __syncthreads();
  }
  if (threadIdx.x == 0) {
    float m = sh0[0] / Lsz;
    float v = sh1[0] / Lsz - m * m;
    stats[2 * nc] = m;
    stats[2 * nc + 1] = rsqrtf(v + 1e-5f);
  }
}

__global__ void inorm_apply_kernel(const float* __restrict__ x,
                                   const float* __restrict__ stats,
                                   float* __restrict__ y, int Lsz, int act,
                                   size_t total) {
  size_t idx = (size_t)blockIdx.x * blockDim.x + threadIdx.x;
  if (idx >= total) return;
  size_t nc = idx / Lsz;
  float v = (x[idx] - stats[2 * nc]) * stats[2 * nc + 1];
  y[idx] = act_apply(v, act);
}

// Layer norm over channel dim for token-major (T, C). One thread per token.
__global__ void layernorm_tok_kernel(const float* __restrict__ x,
                                     const float* __restrict__ w,
                                     const float* __restrict__ b,
                                     float* __restrict__ y, int T, int C) {
  int t = blockIdx.x * blockDim.x + threadIdx.x;
  if (t >= T) return;
  const float* p = x + (size_t)t * C;
  float m = 0.f;
  for (int c = 0; c < C; ++c) m += p[c];
  m /= C;
  float v = 0.f;
  for (int c = 0; c < C; ++c) { float d = p[c] - m; v += d * d; }
  float rs = rsqrtf(v / C + 1e-5f);
  float* q = y + (size_t)t * C;
  for (int c = 0; c < C; ++c) q[c] = (p[c] - m) * rs * w[c] + b[c];
}

// ---------------------------------------------------------------------------
// Layout shuffles & elementwise ops.
// ---------------------------------------------------------------------------
__global__ void bcl_to_blc_kernel(const float* __restrict__ x,
                                  float* __restrict__ y, int C, int L,
                                  size_t total) {
  size_t idx = (size_t)blockIdx.x * blockDim.x + threadIdx.x;
  if (idx >= total) return;
  int l = idx % L; size_t t = idx / L;
  int c = t % C; int b = (int)(t / C);
  y[((size_t)b * L + l) * C + c] = x[idx];
}

__global__ void blc_add_to_bcl_kernel(const float* __restrict__ y,
                                      float* __restrict__ x, int C, int L,
                                      size_t total) {
  size_t idx = (size_t)blockIdx.x * blockDim.x + threadIdx.x;
  if (idx >= total) return;
  int l = idx % L; size_t t = idx / L;
  int c = t % C; int b = (int)(t / C);
  x[idx] += y[((size_t)b * L + l) * C + c];
}

__global__ void slice_cols_kernel(const float* __restrict__ src, int sld,
                                  int soff, float* __restrict__ dst, int cols,
                                  size_t rows) {
  size_t idx = (size_t)blockIdx.x * blockDim.x + threadIdx.x;
  size_t total = rows * cols;
  if (idx >= total) return;
  size_t r = idx / cols; int c = idx % cols;
  dst[idx] = src[r * sld + soff + c];
}

__global__ void flip_mid_kernel(const float* __restrict__ src,
                                float* __restrict__ dst, int L, int D,
                                size_t total) {
  size_t idx = (size_t)blockIdx.x * blockDim.x + threadIdx.x;
  if (idx >= total) return;
  int d = idx % D; size_t t = idx / D;
  int l = t % L; int b = (int)(t / L);
  dst[idx] = src[((size_t)b * L + (L - 1 - l)) * D + d];
}

// Slice shuffle: fwd: us[b, t*ns+s, d] = u[b, s*l2+t, d]; inverse undoes it.
__global__ void gather_tok_kernel(const float* __restrict__ src,
                                  float* __restrict__ dst, int L, int D,
                                  int ns, int l2, int inverse, size_t total) {
  size_t idx = (size_t)blockIdx.x * blockDim.x + threadIdx.x;
  if (idx >= total) return;
  int d = idx % D; size_t t = idx / D;
  int j = t % L; int b = (int)(t / L);
  int s = j % ns, tt = j / ns;
  size_t mapped = ((size_t)b * L + (size_t)s * l2 + tt) * D + d;
  if (!inverse) dst[idx] = src[mapped];
  else          dst[mapped] = src[idx];
}

__global__ void add_kernel(const float* __restrict__ a,
                           const float* __restrict__ b,
                           float* __restrict__ o, size_t n) {
  size_t i = (size_t)blockIdx.x * blockDim.x + threadIdx.x;
  if (i < n) o[i] = a[i] + b[i];
}

__global__ void comb_kernel(const float* __restrict__ yf,
                            const float* __restrict__ yb,
                            const float* __restrict__ ys,
                            const float* __restrict__ z,
                            float* __restrict__ o, size_t n) {
  size_t i = (size_t)blockIdx.x * blockDim.x + threadIdx.x;
  if (i < n) {
    float zz = z[i];
    o[i] = (yf[i] + yb[i] + ys[i]) * (zz / (1.f + __expf(-zz)));
  }
}

__global__ void fuse_gate_kernel(const float* __restrict__ gate,
                                 const float* __restrict__ feat,
                                 const float* __restrict__ style,
                                 float* __restrict__ o, int C, int L,
                                 size_t total) {
  size_t idx = (size_t)blockIdx.x * blockDim.x + threadIdx.x;
  if (idx >= total) return;
  int l = idx % L; size_t t = idx / L;
  int b = (int)(t / C);
  float g = gate[(size_t)b * L + l];
  o[idx] = g * feat[idx] + (1.f - g) * style[idx];
}

// Trilinear resize (half-pixel centers), NC independent channels.
__global__ void resize_tri_kernel(const float* __restrict__ src,
                                  float* __restrict__ dst, int SD, int SH,
                                  int SW, int DD, int DH, int DW,
                                  size_t total) {
  size_t idx = (size_t)blockIdx.x * blockDim.x + threadIdx.x;
  if (idx >= total) return;
  int ow = idx % DW; size_t t = idx / DW;
  int oh = t % DH; t /= DH;
  int od = t % DD; int nc = (int)(t / DD);
  float fd = ((od + 0.5f) * SD) / DD - 0.5f;
  float fh = ((oh + 0.5f) * SH) / DH - 0.5f;
  float fw = ((ow + 0.5f) * SW) / DW - 0.5f;
  int d0 = (int)floorf(fd), h0 = (int)floorf(fh), w0 = (int)floorf(fw);
  float wd = fd - d0, wh = fh - h0, ww = fw - w0;
  int d0c = max(0, min(SD - 1, d0)), d1c = max(0, min(SD - 1, d0 + 1));
  int h0c = max(0, min(SH - 1, h0)), h1c = max(0, min(SH - 1, h0 + 1));
  int w0c = max(0, min(SW - 1, w0)), w1c = max(0, min(SW - 1, w0 + 1));
  const float* p = src + (size_t)nc * SD * SH * SW;
#define SRC(dz, hy, wx) p[((size_t)(dz) * SH + (hy)) * SW + (wx)]
  float c00 = SRC(d0c, h0c, w0c) * (1 - ww) + SRC(d0c, h0c, w1c) * ww;
  float c01 = SRC(d0c, h1c, w0c) * (1 - ww) + SRC(d0c, h1c, w1c) * ww;
  float c10 = SRC(d1c, h0c, w0c) * (1 - ww) + SRC(d1c, h0c, w1c) * ww;
  float c11 = SRC(d1c, h1c, w0c) * (1 - ww) + SRC(d1c, h1c, w1c) * ww;
#undef SRC
  float c0 = c00 * (1 - wh) + c01 * wh;
  float c1 = c10 * (1 - wh) + c11 * wh;
  dst[idx] = c0 * (1 - wd) + c1 * wd;
}

// Depthwise causal conv1d (k=4, left pad 3) + SiLU. u is (B, L, D) token-major.
__global__ void cconv1d_silu_kernel(const float* __restrict__ u,
                                    const float* __restrict__ w,
                                    const float* __restrict__ b,
                                    float* __restrict__ y, int L, int D,
                                    size_t total) {
  size_t idx = (size_t)blockIdx.x * blockDim.x + threadIdx.x;
  if (idx >= total) return;
  int d = idx % D; size_t t = idx / D;
  int l = t % L; int bb = (int)(t / L);
  float s = b[d];
#pragma unroll
  for (int j = 0; j < 4; ++j) {
    int ll = l - 3 + j;
    if (ll >= 0) s += w[d * 4 + j] * u[((size_t)bb * L + ll) * D + d];
  }
  y[idx] = s / (1.f + __expf(-s));
}

// Selective-scan: one thread per (batch, channel); 16 SSM states in VGPRs.
// bc points at the B-part of dbl rows (bc row = [B(16) | C(16)]), ldbc strided.
__global__ void ssm_scan_kernel(const float* __restrict__ u,
                                const float* __restrict__ dt,
                                const float* __restrict__ bc, int ldbc,
                                const float* __restrict__ A_log,
                                const float* __restrict__ Dp,
                                float* __restrict__ y, int B, int L, int Dd) {
  int idx = blockIdx.x * blockDim.x + threadIdx.x;
  if (idx >= B * Dd) return;
  int d = idx % Dd, bb = idx / Dd;
  float An[16], h[16];
#pragma unroll
  for (int s = 0; s < 16; ++s) {
    An[s] = -__expf(A_log[(size_t)d * 16 + s]);
    h[s] = 0.f;
  }
  float Dc = Dp[d];
  for (int l = 0; l < L; ++l) {
    size_t t = (size_t)bb * L + l;
    const float* row = bc + t * ldbc;
    __builtin_prefetch(row + ldbc, 0, 0);  // next token's B/C row
    float ut = u[t * Dd + d];
    float dtt = dt[t * Dd + d];
    float du = dtt * ut;
    float acc = ut * Dc;
#pragma unroll
    for (int s = 0; s < 16; ++s) {
      float dA = __expf(dtt * An[s]);
      h[s] = dA * h[s] + du * row[s];
      acc += h[s] * row[16 + s];
    }
    y[t * Dd + d] = acc;
  }
}

// ---------------------------------------------------------------------------
// Host side
// ---------------------------------------------------------------------------
static inline int cdiv(int a, int b) { return (a + b - 1) / b; }
static inline size_t scdiv(size_t a, size_t b) { return (a + b - 1) / b; }

struct DirP { const float *A_log, *D, *conv_b, *conv_w, *dt_b, *dt_w, *xproj; };
struct LayerP { DirP bwd, fwd, slc; const float *in_proj, *ln_b, *ln_w, *out_proj; };
struct StageP {
  const float *p1b, *p1w, *p2b, *p2w, *p3b, *p3w, *p4b, *p4w;       // gsc
  const float *g1b, *g1w, *g2b, *g2w, *ob, *ow, *pb, *pw;           // him
  LayerP layers[2];
  const float *f1b, *f1w, *f2b, *f2w;                               // mlp
};

static void gemm(hipStream_t st, const float* A, int lda, int a_cm,
                 const float* W, int ldw, const float* bias, float* C, int ldc,
                 int c_cm, const float* res, int M, int N, int K, int act) {
  dim3 grid(cdiv(M, 16), cdiv(N, 128), 1);
  wmma_gemm_kernel<<<grid, 256, 0, st>>>(A, lda, a_cm, W, ldw, bias, C, ldc,
                                         c_cm, res, M, N, K, act);
}

static void inorm(hipStream_t st, const float* x, float* y, float* stats,
                  int N, int C, int Lsz, int act) {
  inorm_stats_kernel<<<N * C, 256, 0, st>>>(x, stats, Lsz);
  size_t total = (size_t)N * C * Lsz;
  inorm_apply_kernel<<<(int)scdiv(total, 256), 256, 0, st>>>(x, stats, y, Lsz,
                                                             act, total);
}

static void dir_run(hipStream_t st, const float* uin, const DirP& dp,
                    float* yout, int B, int L, int di, int R, float* UC,
                    float* DT, float* DBL) {
  int T = B * L;
  size_t n = (size_t)T * di;
  cconv1d_silu_kernel<<<(int)scdiv(n, 256), 256, 0, st>>>(uin, dp.conv_w,
                                                          dp.conv_b, UC, L, di, n);
  // dbl = uc @ xproj.T   ((R+32) x di weight)
  gemm(st, UC, di, 0, dp.xproj, di, nullptr, DBL, R + 32, 0, nullptr, T,
       R + 32, di, ACT_NONE);
  // dt = softplus(dbl[:, :R] @ dt_w.T + dt_b)
  gemm(st, DBL, R + 32, 0, dp.dt_w, R, dp.dt_b, DT, di, 0, nullptr, T, di, R,
       ACT_SOFTPLUS);
  ssm_scan_kernel<<<cdiv(B * di, 128), 128, 0, st>>>(UC, DT, DBL + R, R + 32,
                                                     dp.A_log, dp.D, yout, B,
                                                     L, di);
}

extern "C" void kernel_launch(void* const* d_in, const int* in_sizes, int n_in,
                              void* d_out, int out_size, void* d_ws,
                              size_t ws_size, hipStream_t stream) {
  (void)in_sizes; (void)n_in; (void)out_size; (void)ws_size;
  static const int DIMS_[4] = {48, 96, 192, 384};
  static const int NS_[4]   = {64, 32, 16, 8};
  static const int SS_[4]   = {16, 8, 4, 2};
  const int B = 2;

  const float* x_in    = (const float*)d_in[0];
  const float* x_style = (const float*)d_in[1];

  // ---- decode params (JAX pytree: dict keys sorted) -----------------------
  int pi = 2;
  auto nxt = [&]() { return (const float*)d_in[pi++]; };
  const float *down_b_[3], *down_w_[3];
  for (int i = 0; i < 3; ++i) { down_b_[i] = nxt(); down_w_[i] = nxt(); }
  StageP st[4];
  for (int i = 0; i < 4; ++i) {
    StageP& s = st[i];
    s.p1b = nxt(); s.p1w = nxt(); s.p2b = nxt(); s.p2w = nxt();
    s.p3b = nxt(); s.p3w = nxt(); s.p4b = nxt(); s.p4w = nxt();
    s.g1b = nxt(); s.g1w = nxt(); s.g2b = nxt(); s.g2w = nxt();
    s.ob  = nxt(); s.ow  = nxt(); s.pb  = nxt(); s.pw  = nxt();
    for (int l = 0; l < 2; ++l) {
      LayerP& lp = s.layers[l];
      auto rdir = [&](DirP& d) {
        d.A_log = nxt(); d.D = nxt(); d.conv_b = nxt(); d.conv_w = nxt();
        d.dt_b = nxt(); d.dt_w = nxt(); d.xproj = nxt();
      };
      rdir(lp.bwd); rdir(lp.fwd);
      lp.in_proj = nxt(); lp.ln_b = nxt(); lp.ln_w = nxt(); lp.out_proj = nxt();
      rdir(lp.slc);
    }
    s.f1b = nxt(); s.f1w = nxt(); s.f2b = nxt(); s.f2w = nxt();
  }
  const float* stem_b = nxt();
  const float* stem_w = nxt();

  // ---- workspace bump allocator ------------------------------------------
  char* wp = (char*)d_ws;
  auto alloc = [&](size_t nfloats) {
    float* p = (float*)wp;
    wp += ((nfloats * sizeof(float) + 255) / 256) * 256;
    return p;
  };
  const size_t FEAT = 393216;   // B * 48 * 16^3 (max channel-first feature)
  float* Xa = alloc(FEAT);  float* Xb = alloc(FEAT);
  float* Ta = alloc(FEAT);  float* Tb = alloc(FEAT);  float* Tc = alloc(FEAT);
  float* W2 = alloc(2 * FEAT);           // 2C-channel temp (concat / MLP hidden)
  float* SR = alloc((size_t)B * 64 * 4096);   // resized style
  float* SP = alloc(FEAT);               // projected style
  float* GA = alloc((size_t)B * 4096);   // gate
  float* XF = alloc(FEAT);  float* XN = alloc(FEAT);      // token-major
  float* XZ = alloc(4 * FEAT);           // in_proj output (B,L,4C)
  float* U  = alloc(2 * FEAT); float* Z  = alloc(2 * FEAT);
  float* UC = alloc(2 * FEAT); float* UF = alloc(2 * FEAT);
  float* DT = alloc(2 * FEAT);
  float* YF = alloc(2 * FEAT); float* YB = alloc(2 * FEAT);
  float* YS = alloc(2 * FEAT); float* YT = alloc(2 * FEAT);
  float* DBL = alloc((size_t)B * 4096 * 56);
  float* YBL = alloc(FEAT);
  float* STATS = alloc(2048);

  float* X = Xa;
  float* Xalt = Xb;
  float* dout = (float*)d_out;
  size_t out_off = 0;

  // ---- stem: (2,1,32,32,32) -> (2,48,16,16,16) ---------------------------
  {
    size_t total = (size_t)B * 48 * 16 * 16 * 16;
    conv3d_kernel<<<(int)scdiv(total, 256), 256, 0, stream>>>(
        x_in, stem_w, stem_b, X, B, 1, 32, 32, 32, 48, 7, 2, 3, 16, 16, 16);
  }

  int prevC = 48, prevS = 16;
  for (int i = 0; i < 4; ++i) {
    const int C = DIMS_[i], S = SS_[i];
    const int L = S * S * S;
    const int di = 2 * C;
    const int R = (C + 15) / 16;     // dt_rank = ceil(C/16)
    const int ns = NS_[i], l2 = L / ns;
    const StageP& sp = st[i];
    const size_t FL = (size_t)B * C * L;     // channel-first feature elements
    const size_t TL = (size_t)B * L;         // tokens

    if (i > 0) {  // down transition: inorm -> 2x2x2 stride-2 conv
      inorm(stream, X, Ta, STATS, B, prevC, prevS * prevS * prevS, ACT_NONE);
      conv3d_kernel<<<(int)scdiv(FL, 256), 256, 0, stream>>>(
          Ta, down_w_[i - 1], down_b_[i - 1], Xalt, B, prevC, prevS, prevS,
          prevS, C, 2, 2, 0, S, S, S);
      float* t = X; X = Xalt; Xalt = t;
    }

    // ---- gsc ------------------------------------------------------------
    {
      conv3d_kernel<<<(int)scdiv(FL, 256), 256, 0, stream>>>(
          X, sp.p1w, sp.p1b, Ta, B, C, S, S, S, C, 3, 1, 1, S, S, S);
      inorm(stream, Ta, Ta, STATS, B, C, L, ACT_RELU);
      conv3d_kernel<<<(int)scdiv(FL, 256), 256, 0, stream>>>(
          Ta, sp.p2w, sp.p2b, Tb, B, C, S, S, S, C, 3, 1, 1, S, S, S);
      inorm(stream, Tb, Tb, STATS, B, C, L, ACT_RELU);          // x1
      for (int n = 0; n < B; ++n)                                // 1x1 conv p3
        gemm(stream, X + (size_t)n * C * L, L, 1, sp.p3w, C, sp.p3b,
             Tc + (size_t)n * C * L, L, 1, nullptr, L, C, C, ACT_NONE);
      inorm(stream, Tc, Tc, STATS, B, C, L, ACT_RELU);          // x2
      add_kernel<<<(int)scdiv(FL, 256), 256, 0, stream>>>(Tb, Tc, Ta, FL);
      for (int n = 0; n < B; ++n)                                // 1x1 conv p4
        gemm(stream, Ta + (size_t)n * C * L, L, 1, sp.p4w, C, sp.p4b,
             Tb + (size_t)n * C * L, L, 1, nullptr, L, C, C, ACT_NONE);
      inorm(stream, Tb, Tb, STATS, B, C, L, ACT_RELU);
      add_kernel<<<(int)scdiv(FL, 256), 256, 0, stream>>>(Tb, X, Xalt, FL);
      float* t = X; X = Xalt; Xalt = t;
    }

    // ---- him ------------------------------------------------------------
    {
      size_t srl = (size_t)B * 64 * L;
      resize_tri_kernel<<<(int)scdiv(srl, 256), 256, 0, stream>>>(
          x_style, SR, 16, 16, 16, S, S, S, srl);
      for (int n = 0; n < B; ++n)                                // style proj
        gemm(stream, SR + (size_t)n * 64 * L, L, 1, sp.pw, 64, sp.pb,
             SP + (size_t)n * C * L, L, 1, nullptr, L, C, 64, ACT_NONE);
      for (int n = 0; n < B; ++n) {                              // concat
        hipMemcpyAsync(W2 + (size_t)n * 2 * C * L, X + (size_t)n * C * L,
                       (size_t)C * L * 4, hipMemcpyDeviceToDevice, stream);
        hipMemcpyAsync(W2 + (size_t)n * 2 * C * L + (size_t)C * L,
                       SP + (size_t)n * C * L, (size_t)C * L * 4,
                       hipMemcpyDeviceToDevice, stream);
      }
      for (int n = 0; n < B; ++n)                                // g1 + relu
        gemm(stream, W2 + (size_t)n * 2 * C * L, L, 1, sp.g1w, 2 * C, sp.g1b,
             Ta + (size_t)n * C * L, L, 1, nullptr, L, C, 2 * C, ACT_RELU);
      for (int n = 0; n < B; ++n)                                // g2 + sigmoid
        gemm(stream, Ta + (size_t)n * C * L, L, 1, sp.g2w, C, sp.g2b,
             GA + (size_t)n * L, L, 1, nullptr, L, 1, C, ACT_SIGMOID);
      fuse_gate_kernel<<<(int)scdiv(FL, 256), 256, 0, stream>>>(GA, X, SP, Tb,
                                                                C, L, FL);
      for (int n = 0; n < B; ++n)                                // ow + resid
        gemm(stream, Tb + (size_t)n * C * L, L, 1, sp.ow, C, sp.ob,
             Xalt + (size_t)n * C * L, L, 1, X + (size_t)n * C * L, L, C, C,
             ACT_NONE);
      float* t = X; X = Xalt; Xalt = t;
    }

    // ---- mamba layers ---------------------------------------------------
    for (int ly = 0; ly < 2; ++ly) {
      const LayerP& lp = sp.layers[ly];
      bcl_to_blc_kernel<<<(int)scdiv(FL, 256), 256, 0, stream>>>(X, XF, C, L, FL);
      layernorm_tok_kernel<<<(int)scdiv(TL, 256), 256, 0, stream>>>(
          XF, lp.ln_w, lp.ln_b, XN, (int)TL, C);
      // xz = xn @ in_proj.T : (B*L, 4C)
      gemm(stream, XN, C, 0, lp.in_proj, C, nullptr, XZ, 4 * C, 0, nullptr,
           (int)TL, 4 * C, C, ACT_NONE);
      slice_cols_kernel<<<(int)scdiv(TL * di, 256), 256, 0, stream>>>(
          XZ, 4 * C, 0, U, di, TL);
      slice_cols_kernel<<<(int)scdiv(TL * di, 256), 256, 0, stream>>>(
          XZ, 4 * C, di, Z, di, TL);
      size_t n = TL * di;
      // forward
      dir_run(stream, U, lp.fwd, YF, B, L, di, R, UC, DT, DBL);
      // backward (flip, run, flip)
      flip_mid_kernel<<<(int)scdiv(n, 256), 256, 0, stream>>>(U, UF, L, di, n);
      dir_run(stream, UF, lp.bwd, YT, B, L, di, R, UC, DT, DBL);
      flip_mid_kernel<<<(int)scdiv(n, 256), 256, 0, stream>>>(YT, YB, L, di, n);
      // slice-shuffled
      gather_tok_kernel<<<(int)scdiv(n, 256), 256, 0, stream>>>(U, UF, L, di,
                                                                ns, l2, 0, n);
      dir_run(stream, UF, lp.slc, YT, B, L, di, R, UC, DT, DBL);
      gather_tok_kernel<<<(int)scdiv(n, 256), 256, 0, stream>>>(YT, YS, L, di,
                                                                ns, l2, 1, n);
      // combine * silu(z), out_proj, residual back into channel-first X
      comb_kernel<<<(int)scdiv(n, 256), 256, 0, stream>>>(YF, YB, YS, Z, YT, n);
      gemm(stream, YT, di, 0, lp.out_proj, di, nullptr, YBL, C, 0, nullptr,
           (int)TL, C, di, ACT_NONE);
      blc_add_to_bcl_kernel<<<(int)scdiv(FL, 256), 256, 0, stream>>>(YBL, X, C,
                                                                     L, FL);
    }

    // ---- stage output MLP ----------------------------------------------
    {
      inorm(stream, X, Ta, STATS, B, C, L, ACT_NONE);
      for (int n = 0; n < B; ++n)                                // f1 + gelu
        gemm(stream, Ta + (size_t)n * C * L, L, 1, sp.f1w, C, sp.f1b,
             W2 + (size_t)n * 2 * C * L, L, 1, nullptr, L, 2 * C, C, ACT_GELU);
      for (int n = 0; n < B; ++n)                                // f2 -> d_out
        gemm(stream, W2 + (size_t)n * 2 * C * L, L, 1, sp.f2w, 2 * C, sp.f2b,
             dout + out_off + (size_t)n * C * L, L, 1, nullptr, L, C, 2 * C,
             ACT_NONE);
      out_off += (size_t)B * C * L;
    }

    prevC = C;
    prevS = S;
  }
}
